// GCNNet_8005819040454
// MI455X (gfx1250) — matrix operand order repaired and checked by reading, not compiled, not run on hardware
//
#include <hip/hip_runtime.h>

typedef _Float16 half_t;
typedef __attribute__((ext_vector_type(16))) _Float16 v16h;
typedef __attribute__((ext_vector_type(8)))  float    v8f;
typedef __attribute__((ext_vector_type(4)))  unsigned int u32x4;
typedef __attribute__((ext_vector_type(8)))  int      i32x8;
typedef __attribute__((ext_vector_type(4)))  int      i32x4;

#define N_NODES 40000
#define N_EDGES 640000
#define K_DIM   128

// ------------------------- degree / normalization -------------------------

__global__ void deg_init(float* __restrict__ deg, int n) {
    int i = blockIdx.x * blockDim.x + threadIdx.x;
    if (i < n) deg[i] = 1.0f;  // self-loop contribution
}

__global__ void deg_accum(const int* __restrict__ dst, float* __restrict__ deg, int nE) {
    int e = blockIdx.x * blockDim.x + threadIdx.x;
    if (e < nE) atomicAdd(&deg[dst[e]], 1.0f);
}

__global__ void deg_rsqrt(const float* __restrict__ deg, float* __restrict__ dinv, int n) {
    int i = blockIdx.x * blockDim.x + threadIdx.x;
    if (i < n) dinv[i] = rsqrtf(deg[i]);
}

// ------------------------- f32 -> f16 feature conversion -------------------------

__global__ void cvt_f16(const float* __restrict__ w, half_t* __restrict__ wh, int n) {
    int i = blockIdx.x * blockDim.x + threadIdx.x;
    if (i < n) wh[i] = (half_t)w[i];
}

// ------------- pack W (KxN f32, row-major) into WMMA B fragment layout -------------
// Packed: Bp[((ntile*4 + kstep)*32 + lane)*16 + h], with k = kstep*32 + 16*(lane>>4) + h,
// n = nt*16 + (lane&15).  Each lane's fragment = 32 contiguous bytes -> 2x b128 loads.

template<int NCOLS>
__global__ void pack_b(const float* __restrict__ W, half_t* __restrict__ Bp) {
    int i = blockIdx.x * blockDim.x + threadIdx.x;
    if (i >= NCOLS * 128) return;
    int h    = i & 15;
    int lane = (i >> 4) & 31;
    int ks   = (i >> 9) & 3;
    int nt   = i >> 11;
    int n = nt * 16 + (lane & 15);
    int k = ks * 32 + 16 * (lane >> 4) + h;
    Bp[i] = (half_t)W[(size_t)k * NCOLS + n];
}

// ------------------------- WMMA GEMM + fused self-loop term -------------------------
// X16 is f16 row-major [N_NODES][128]. A tile staged to LDS by the Tensor Data Mover
// (2D D#, 2-byte elements, HW padding 4 DWORDs per 64-DWORD row -> [16][136] layout).
// HW = X @ W (f32) and AG = HW * dinv^2 (self-loop init of the aggregation buffer).

template<int NCOLS>
__global__ void __launch_bounds__(NCOLS * 2)
gemm_wmma(const half_t* __restrict__ X16, const half_t* __restrict__ Bpack,
          const float* __restrict__ dinv, float* __restrict__ HW, float* __restrict__ AG) {
    __shared__ half_t As[16][K_DIM + 8];   // +8 halfs pad (matches TDM pad_amount)

    const int m0  = blockIdx.x * 16;
    const int tid = threadIdx.x;

#if __has_builtin(__builtin_amdgcn_tensor_load_to_lds) && \
    __has_builtin(__builtin_amdgcn_s_wait_tensorcnt)
    if (tid < 32) {   // wave 0 issues the TDM transfer (EXEC ignored by TDM)
        unsigned long long gaddr =
            (unsigned long long)(const void*)(X16 + (size_t)m0 * K_DIM);
        u32x4 g0;
        g0[0] = 1u;                                        // count=1, user descriptor
        g0[1] = (unsigned int)(unsigned long long)(void*)&As[0][0];  // lds_addr
        g0[2] = (unsigned int)gaddr;                       // global_addr[31:0]
        g0[3] = (unsigned int)((gaddr >> 32) & 0x01FFFFFFu) | (2u << 30);  // addr hi, type=2
        i32x8 g1;
        g1[0] = (int)((1u << 16)      // data_size = 2 bytes
                    | (1u << 20)      // pad_enable
                    | (5u << 22)      // pad_interval: 64 DWORDs (one 128-half row)
                    | (3u << 25));    // pad_amount: 4 DWORDs (8 halfs)
        g1[1] = (int)(128u << 16);    // tensor_dim0 = 128
        g1[2] = (int)(16u << 16);     // tensor_dim1 = 16
        g1[3] = (int)(128u << 16);    // tile_dim0 = 128
        g1[4] = (int)16u;             // tile_dim1 = 16, tile_dim2 = 0
        g1[5] = (int)128u;            // tensor_dim0_stride = 128
        g1[6] = 0;
        g1[7] = 0;
        i32x4 z4 = {0, 0, 0, 0};
#if __clang_major__ >= 23
        i32x8 z8 = {0, 0, 0, 0, 0, 0, 0, 0};
        __builtin_amdgcn_tensor_load_to_lds(g0, g1, z4, z4, z8, 0);
#else
        __builtin_amdgcn_tensor_load_to_lds(g0, g1, z4, z4, 0);
#endif
        __builtin_amdgcn_s_wait_tensorcnt(0);
    }
#else
    for (int i = tid; i < 16 * K_DIM; i += NCOLS * 2) {
        int r = i >> 7, k = i & 127;
        As[r][k] = X16[(size_t)(m0 + r) * K_DIM + k];
    }
#endif
    __syncthreads();

    const int lane  = tid & 31;
    const int ntile = tid >> 5;        // wave id = output column tile
    const int kg    = lane >> 4;       // lane half-group
    const int rl    = lane & 15;       // row (A) / column (B, C/D)
    const int nbase = ntile * 16 + rl;

    const v16h* Bp = (const v16h*)Bpack;

    v8f c = {};
    #pragma unroll
    for (int ks = 0; ks < 4; ++ks) {
        const int k0 = ks * 32;
        v16h a;
        #pragma unroll
        for (int j = 0; j < 8; ++j) {
            int ka = 2 * j + 16 * kg;
            a[2 * j]     = As[rl][k0 + ka];
            a[2 * j + 1] = As[rl][k0 + ka + 1];
        }
        v16h b = Bp[(ntile * 4 + ks) * 32 + lane];
        c = __builtin_amdgcn_wmma_f32_16x16x32_f16(false, a, false, b, (short)0, c,
                                                   false, false);
    }

    #pragma unroll
    for (int r = 0; r < 8; ++r) {
        int   m   = m0 + r + 8 * kg;
        float val = c[r];
        float di  = dinv[m];
        HW[(size_t)m * NCOLS + nbase] = val;             // for edge gather
        AG[(size_t)m * NCOLS + nbase] = val * di * di;   // fused self-loop init
    }
}

// ------------------------- edge scatter: one wave32 per edge -------------------------
// Edge index is wave-uniform -> force scalar so src/dst/dinv fetches become SMEM loads.

template<int D>
__global__ void edge_agg(const int* __restrict__ src, const int* __restrict__ dst,
                         const float* __restrict__ dinv, const float* __restrict__ hw,
                         float* __restrict__ agg, int nE) {
    const int gid  = blockIdx.x * blockDim.x + threadIdx.x;
    const int lane = gid & 31;
    int e = gid >> 5;
    if (e >= nE) return;
    e = __builtin_amdgcn_readfirstlane(e);

    const int   s   = src[e];
    const int   d   = dst[e];
    const float nrm = dinv[s] * dinv[d];

    constexpr int CH = D / 32;   // floats per lane (4 for D=128, 2 for D=64)
    const float* hs = hw  + (size_t)s * D + lane * CH;
    float*       ad = agg + (size_t)d * D + lane * CH;

    float v[CH];
    #pragma unroll
    for (int c = 0; c < CH; ++c) v[c] = hs[c];     // contiguous -> b128/b64 load
    #pragma unroll
    for (int c = 0; c < CH; ++c) atomicAdd(&ad[c], v[c] * nrm);
}

// ------------------------- bias + (optional) leaky-relu -------------------------

template<int D, bool ACT, typename OT>
__global__ void post_act(const float* __restrict__ agg, const float* __restrict__ bias,
                         OT* __restrict__ out, int n) {
    int i = blockIdx.x * blockDim.x + threadIdx.x;
    if (i >= n * D) return;
    float v = agg[i] + bias[i & (D - 1)];
    if (ACT) v = (v > 0.0f) ? v : 0.01f * v;
    out[i] = (OT)v;
}

// ------------------------- host side -------------------------

extern "C" void kernel_launch(void* const* d_in, const int* in_sizes, int n_in,
                              void* d_out, int out_size, void* d_ws, size_t ws_size,
                              hipStream_t stream) {
    const float* x   = (const float*)d_in[0];
    const int*   ei  = (const int*)d_in[1];
    const int*   src = ei;
    const int*   dst = ei + N_EDGES;
    const float* W1  = (const float*)d_in[2];
    const float* b1  = (const float*)d_in[3];
    const float* W2  = (const float*)d_in[4];
    const float* b2  = (const float*)d_in[5];
    const float* W3  = (const float*)d_in[6];
    const float* b3  = (const float*)d_in[7];
    float* out = (float*)d_out;

    // workspace layout (bytes)
    char* ws = (char*)d_ws;
    const size_t FEAT_BYTES  = (size_t)N_NODES * 128 * sizeof(float);    // 20.48 MB
    const size_t FEAT16_BYTES = (size_t)N_NODES * 128 * sizeof(half_t);  // 10.24 MB
    float*  bufW  = (float*)(ws);                                        // gemm output hw (f32)
    float*  bufA  = (float*)(ws + FEAT_BYTES);                           // aggregation (f32)
    half_t* bufX16 = (half_t*)(ws + 2 * FEAT_BYTES);                     // x in f16
    half_t* bufH16 = (half_t*)(ws + 2 * FEAT_BYTES + FEAT16_BYTES);      // hidden in f16
    float*  deg   = (float*)(ws + 2 * FEAT_BYTES + 2 * FEAT16_BYTES);
    float*  dinv  = (float*)(ws + 2 * FEAT_BYTES + 2 * FEAT16_BYTES + (size_t)N_NODES * 4);
    half_t* w1p   = (half_t*)(ws + 2 * FEAT_BYTES + 2 * FEAT16_BYTES + (size_t)2 * N_NODES * 4);
    half_t* w2p   = w1p + 128 * 128;
    half_t* w3p   = w2p + 128 * 128;

    const int T = 256;
    const int gN128 = (N_NODES * 128 + T - 1) / T;
    const int gN64  = (N_NODES * 64  + T - 1) / T;
    const int gE    = (N_EDGES + T - 1) / T;
    const int gE32  = (N_EDGES * 32 + T - 1) / T;   // wave per edge
    const int gNode = (N_NODES + T - 1) / T;

    // normalization coefficients
    deg_init<<<gNode, T, 0, stream>>>(deg, N_NODES);
    deg_accum<<<gE, T, 0, stream>>>(dst, deg, N_EDGES);
    deg_rsqrt<<<gNode, T, 0, stream>>>(deg, dinv, N_NODES);

    // weights -> packed f16 WMMA B fragments; x -> f16
    pack_b<128><<<(128 * 128 + T - 1) / T, T, 0, stream>>>(W1, w1p);
    pack_b<128><<<(128 * 128 + T - 1) / T, T, 0, stream>>>(W2, w2p);
    pack_b<64><<<(64 * 128 + T - 1) / T, T, 0, stream>>>(W3, w3p);
    cvt_f16<<<gN128, T, 0, stream>>>(x, bufX16, N_NODES * 128);

    // ---- layer 1: x16 -> h16 ----
    gemm_wmma<128><<<N_NODES / 16, 256, 0, stream>>>(bufX16, w1p, dinv, bufW, bufA);
    edge_agg<128><<<gE32, T, 0, stream>>>(src, dst, dinv, bufW, bufA, N_EDGES);
    post_act<128, true, half_t><<<gN128, T, 0, stream>>>(bufA, b1, bufH16, N_NODES);

    // ---- layer 2: h16 -> h16 ----
    gemm_wmma<128><<<N_NODES / 16, 256, 0, stream>>>(bufH16, w2p, dinv, bufW, bufA);
    edge_agg<128><<<gE32, T, 0, stream>>>(src, dst, dinv, bufW, bufA, N_EDGES);
    post_act<128, true, half_t><<<gN128, T, 0, stream>>>(bufA, b2, bufH16, N_NODES);

    // ---- layer 3: h16 -> d_out (f32) ----
    gemm_wmma<64><<<N_NODES / 16, 128, 0, stream>>>(bufH16, w3p, dinv, bufW, bufA);
    edge_agg<64><<<gE32, T, 0, stream>>>(src, dst, dinv, bufW, bufA, N_EDGES);
    post_act<64, false, float><<<gN64, T, 0, stream>>>(bufA, b3, out, N_NODES);
}
